// VQ_VAE_17394617549039
// MI455X (gfx1250) — matrix operand order repaired and checked by reading, not compiled, not run on hardware
//
#include <hip/hip_runtime.h>
#include <stdint.h>

// ---------------------------------------------------------------------------
// CDNA5 (gfx1250) types for WMMA
// ---------------------------------------------------------------------------
typedef __attribute__((ext_vector_type(16))) __bf16 v16bf;
typedef __attribute__((ext_vector_type(8)))  float  v8f;

union FragBF { uint4 q[2]; v16bf v; };

__device__ __forceinline__ unsigned short f2bf(float x) {
  union { float f; unsigned u; } c; c.f = x;
  unsigned u = c.u;
  u += 0x7FFFu + ((u >> 16) & 1u);   // round-to-nearest-even
  return (unsigned short)(u >> 16);
}

// ---------------------------------------------------------------------------
// Direct conv k=4 s=2 p=1 (small-channel layers; conv bias cancels in train-BN)
// ---------------------------------------------------------------------------
__global__ __launch_bounds__(256)
void conv_k4s2(const float* __restrict__ in, const float* __restrict__ w,
               float* __restrict__ out, int B, int Cin, int Cout, int Hin, int Win) {
  const int Hout = Hin >> 1, Wout = Win >> 1;
  long i = (long)blockIdx.x * blockDim.x + threadIdx.x;
  long total = (long)B * Cout * Hout * Wout;
  if (i >= total) return;
  int ow = (int)(i % Wout); long t = i / Wout;
  int oh = (int)(t % Hout); t /= Hout;
  int co = (int)(t % Cout); int b = (int)(t / Cout);
  float acc = 0.f;
  for (int ci = 0; ci < Cin; ++ci) {
    const float* ip = in + ((size_t)(b * Cin + ci) * Hin) * Win;
    const float* wp = w + (((size_t)co * Cin + ci) << 4);
#pragma unroll
    for (int kh = 0; kh < 4; ++kh) {
      int ih = 2 * oh + kh - 1;
      if ((unsigned)ih >= (unsigned)Hin) continue;
#pragma unroll
      for (int kw = 0; kw < 4; ++kw) {
        int iw = 2 * ow + kw - 1;
        if ((unsigned)iw >= (unsigned)Win) continue;
        acc = fmaf(ip[(size_t)ih * Win + iw], wp[kh * 4 + kw], acc);
      }
    }
  }
  out[i] = acc;
}

// ---------------------------------------------------------------------------
// BatchNorm training-mode statistics (one block per channel, deterministic tree)
// ---------------------------------------------------------------------------
__global__ __launch_bounds__(256)
void bn_stats(const float* __restrict__ y, float* __restrict__ mean,
              float* __restrict__ var, int C, int HW, int B) {
  __shared__ float s1[256], s2[256];
  const int c = blockIdx.x;
  const long plane = (long)B * HW;
  float a = 0.f, q = 0.f;
  for (long t = threadIdx.x; t < plane; t += blockDim.x) {
    long bb = t / HW; long sp = t - bb * HW;
    float v = y[((size_t)bb * C + c) * HW + sp];
    a += v; q += v * v;
  }
  s1[threadIdx.x] = a; s2[threadIdx.x] = q; __syncthreads();
  for (int off = 128; off; off >>= 1) {
    if (threadIdx.x < off) { s1[threadIdx.x] += s1[threadIdx.x + off];
                             s2[threadIdx.x] += s2[threadIdx.x + off]; }
    __syncthreads();
  }
  if (threadIdx.x == 0) {
    float m = s1[0] / (float)plane;
    mean[c] = m;
    var[c]  = s2[0] / (float)plane - m * m;
  }
}

// BN apply + activation in place (mode 0 = relu, 1 = tanh). Grid covers exactly n.
__global__ __launch_bounds__(256)
void bn_act(float* __restrict__ y, const float* __restrict__ g,
            const float* __restrict__ be, const float* __restrict__ mean,
            const float* __restrict__ var, int C, int HW, int tanh_mode) {
  size_t i = (size_t)blockIdx.x * blockDim.x + threadIdx.x;
  int c = (int)((i / (size_t)HW) % (size_t)C);
  float v = (y[i] - mean[c]) * rsqrtf(var[c] + 1e-5f) * g[c] + be[c];
  y[i] = tanh_mode ? tanhf(v) : fmaxf(v, 0.f);
}

// ---------------------------------------------------------------------------
// im2col for conv3 (h2 [64,32,64,64] -> A [65536,512] bf16), k=(ci,kh,kw)
// 8 elements per thread -> b128 stores
// ---------------------------------------------------------------------------
__global__ __launch_bounds__(256)
void im2col3(const float* __restrict__ h2, unsigned short* __restrict__ A) {
  size_t i = (size_t)blockIdx.x * blockDim.x + threadIdx.x;   // < 4194304
  size_t base = i << 3;
  int k0 = (int)(base & 511); size_t m = base >> 9;
  int ow = (int)(m & 31), oh = (int)((m >> 5) & 31), b = (int)(m >> 10);
  union { unsigned short u[8]; uint4 q; } pk;
#pragma unroll
  for (int j = 0; j < 8; ++j) {
    int k = k0 + j;
    int kw = k & 3, kh = (k >> 2) & 3, ci = k >> 4;
    int ih = 2 * oh + kh - 1, iw = 2 * ow + kw - 1;
    float v = 0.f;
    if ((unsigned)ih < 64u && (unsigned)iw < 64u)
      v = h2[(((size_t)b * 32 + ci) * 64 + ih) * 64 + iw];
    pk.u[j] = f2bf(v);
  }
  *(uint4*)(A + base) = pk.q;
}

// contiguous f32 -> bf16, 8/thread
__global__ __launch_bounds__(256)
void cvt_bf16(const float* __restrict__ in, unsigned short* __restrict__ out) {
  size_t i = (size_t)blockIdx.x * blockDim.x + threadIdx.x;
  size_t base = i << 3;
  const float4* p = (const float4*)(in + base);
  float4 a = p[0], b = p[1];
  union { unsigned short u[8]; uint4 q; } pk;
  pk.u[0] = f2bf(a.x); pk.u[1] = f2bf(a.y); pk.u[2] = f2bf(a.z); pk.u[3] = f2bf(a.w);
  pk.u[4] = f2bf(b.x); pk.u[5] = f2bf(b.y); pk.u[6] = f2bf(b.z); pk.u[7] = f2bf(b.w);
  *(uint4*)(out + base) = pk.q;
}

// z_e NCHW [64,512,32,32] -> z_flat [65536,512] bf16 (m = b*1024 + h*32 + w)
__global__ __launch_bounds__(256)
void ze_to_flat_bf16(const float* __restrict__ ze, unsigned short* __restrict__ out) {
  size_t i = (size_t)blockIdx.x * blockDim.x + threadIdx.x;   // < 4194304
  size_t base = i << 3;
  int d0 = (int)(base & 511); size_t m = base >> 9;
  int hw = (int)(m & 1023); int b = (int)(m >> 10);
  const float* src = ze + ((size_t)b * 512 + d0) * 1024 + hw;
  union { unsigned short u[8]; uint4 q; } pk;
#pragma unroll
  for (int j = 0; j < 8; ++j) pk.u[j] = f2bf(src[(size_t)j * 1024]);
  *(uint4*)(out + base) = pk.q;
}

// ---------------------------------------------------------------------------
// bf16 WMMA GEMM, K fixed at 512: C[M,N] f32 = A[M,512] * B[N,512]^T
// Block tile 128x64, 8 waves each 32x32 (2x2 v_wmma_f32_16x16x32_bf16).
// Software-pipelined: next K-panel global b128 loads overlap current WMMAs.
// cmode 0: C row-major; cmode 1: latent-NCHW store [(b*N+n)*1024 + hw]
// ---------------------------------------------------------------------------
__global__ __launch_bounds__(256)
void gemm_bf16_wmma(const unsigned short* __restrict__ A,
                    const unsigned short* __restrict__ Bm,
                    float* __restrict__ C, int M, int N, int cmode) {
  constexpr int Kd = 512;
  __shared__ unsigned short At[128 * 40];   // +8 pad per row
  __shared__ unsigned short Bt[64 * 40];
  const int tid  = threadIdx.x;
  const int m0   = blockIdx.y * 128;
  const int n0   = blockIdx.x * 64;
  const int wave = tid >> 5, lane = tid & 31;
  const int mbase = (wave & 3) * 32, nbase = (wave >> 2) * 32;
  const int mrow = lane & 15, hi = lane >> 4;

  // staging assignment: each thread owns three 16B segments per K-panel
  const int row0 = tid >> 2;              // 0..63
  const int segk = (tid & 3) * 8;         // 0,8,16,24
  const unsigned short* gA0 = A  + ((size_t)(m0 + row0)      << 9) + segk;
  const unsigned short* gA1 = A  + ((size_t)(m0 + row0 + 64) << 9) + segk;
  const unsigned short* gB  = Bm + ((size_t)(n0 + row0)      << 9) + segk;
  unsigned short* lA0 = &At[row0 * 40 + segk];
  unsigned short* lA1 = &At[(row0 + 64) * 40 + segk];
  unsigned short* lB  = &Bt[row0 * 40 + segk];

  // prologue: panel 0 in flight
  uint4 ra0 = *(const uint4*)gA0;
  uint4 ra1 = *(const uint4*)gA1;
  uint4 rb  = *(const uint4*)gB;

  v8f acc[2][2] = {};

  for (int k0 = 0; k0 < Kd; k0 += 32) {
    __syncthreads();                       // prior-panel LDS reads done
    *(uint4*)lA0 = ra0;
    *(uint4*)lA1 = ra1;
    *(uint4*)lB  = rb;
    __syncthreads();
    if (k0 + 32 < Kd) {                    // prefetch next panel during compute
      ra0 = *(const uint4*)(gA0 + k0 + 32);
      ra1 = *(const uint4*)(gA1 + k0 + 32);
      rb  = *(const uint4*)(gB  + k0 + 32);
      __builtin_prefetch(gA0 + k0 + 64, 0, 1);   // global_prefetch_b8
    }

    // fragments per documented 16-bit layouts (A 16x32, B 32x16), b128 LDS reads
    FragBF af[2], bf[2];
#pragma unroll
    for (int mi = 0; mi < 2; ++mi) {
      int ar = mbase + mi * 16 + mrow;
      af[mi].q[0] = *(const uint4*)&At[ar * 40 + hi * 8];        // K = hi*8 .. +7
      af[mi].q[1] = *(const uint4*)&At[ar * 40 + 16 + hi * 8];   // K = 16+hi*8 ..
    }
#pragma unroll
    for (int ni = 0; ni < 2; ++ni) {
      int br = nbase + ni * 16 + mrow;                           // N column = mrow
      bf[ni].q[0] = *(const uint4*)&Bt[br * 40 + hi * 8];        // K = hi*8 .. +7
      bf[ni].q[1] = *(const uint4*)&Bt[br * 40 + 16 + hi * 8];   // K = 16+hi*8 ..
    }
#pragma unroll
    for (int mi = 0; mi < 2; ++mi)
#pragma unroll
      for (int ni = 0; ni < 2; ++ni)
        acc[mi][ni] = __builtin_amdgcn_wmma_f32_16x16x32_bf16(
            false, af[mi].v, false, bf[ni].v, (short)0, acc[mi][ni], false, false);
  }

  // store: C/D layout — col = lane&15, row = r + (lane>>4)*8
#pragma unroll
  for (int mi = 0; mi < 2; ++mi)
#pragma unroll
    for (int ni = 0; ni < 2; ++ni) {
      int col = n0 + nbase + ni * 16 + mrow;
#pragma unroll
      for (int r = 0; r < 8; ++r) {
        long mr = (long)m0 + mbase + mi * 16 + hi * 8 + r;
        size_t off;
        if (cmode == 0) off = (size_t)mr * N + col;
        else { long b = mr >> 10; int hw = (int)(mr & 1023);
               off = ((size_t)b * N + col) * 1024 + hw; }
        C[off] = acc[mi][ni][r];
      }
    }
}

// ---------------------------------------------------------------------------
// VQ: codebook norms, argmin over (||cb_n||^2 - 2*G[m][n]), loss reduction
// ---------------------------------------------------------------------------
__global__ void cb_norm(const float* __restrict__ cb, float* __restrict__ out) {
  int n = blockIdx.x * blockDim.x + threadIdx.x;
  if (n >= 512) return;
  float s = 0.f;
  for (int d = 0; d < 512; ++d) { float v = cb[(size_t)n * 512 + d]; s += v * v; }
  out[n] = s;
}

__global__ __launch_bounds__(256)
void vq_argmin(const float* __restrict__ G, const float* __restrict__ cbn,
               int* __restrict__ idx, float* __restrict__ idx_f) {
  const int wave = threadIdx.x >> 5, lane = threadIdx.x & 31;
  const size_t m = (size_t)blockIdx.x * 8 + wave;
  const float* g = G + m * 512;
  float best = 3.4e38f; int bi = 0;
#pragma unroll
  for (int i = 0; i < 16; ++i) {
    int n = i * 32 + lane;
    float v = cbn[n] - 2.f * g[n];
    if (v < best) { best = v; bi = n; }
  }
  for (int off = 16; off >= 1; off >>= 1) {   // wave32 butterfly, lexicographic min
    float ov = __shfl_xor(best, off);
    int   oi = __shfl_xor(bi,   off);
    if (ov < best || (ov == best && oi < bi)) { best = ov; bi = oi; }
  }
  if (lane == 0) { idx[m] = bi; idx_f[m] = (float)bi; }
}

__global__ __launch_bounds__(256)
void vq_loss(const float* __restrict__ ze, const float* __restrict__ cb,
             const int* __restrict__ idx, float* __restrict__ partial) {
  __shared__ float red[256];
  const size_t total = (size_t)65536 * 512;
  float s = 0.f;
  for (size_t e = (size_t)blockIdx.x * blockDim.x + threadIdx.x; e < total;
       e += (size_t)gridDim.x * blockDim.x) {
    int d = (int)(e & 511); size_t m = e >> 9;
    int hw = (int)(m & 1023); int b = (int)(m >> 10);
    float zq = cb[(size_t)idx[m] * 512 + d];
    float zv = ze[((size_t)b * 512 + d) * 1024 + hw];
    float df = zq - zv;
    s += df * df;
  }
  red[threadIdx.x] = s; __syncthreads();
  for (int off = 128; off; off >>= 1) {
    if (threadIdx.x < off) red[threadIdx.x] += red[threadIdx.x + off];
    __syncthreads();
  }
  if (threadIdx.x == 0) partial[blockIdx.x] = red[0];
}

__global__ __launch_bounds__(256)
void loss_final(const float* __restrict__ partial, float* __restrict__ out) {
  __shared__ float red[256];
  float s = 0.f;
  for (int i = threadIdx.x; i < 2048; i += 256) s += partial[i];
  red[threadIdx.x] = s; __syncthreads();
  for (int off = 128; off; off >>= 1) {
    if (threadIdx.x < off) red[threadIdx.x] += red[threadIdx.x + off];
    __syncthreads();
  }
  if (threadIdx.x == 0) out[0] = 2.f * red[0] / (65536.f * 512.f);
}

// ---------------------------------------------------------------------------
// Transposed conv k=4 s=2 p=1 (gather form: exactly 2x2 taps per output)
// Weight layout torch ConvTranspose: [Cin, Cout, 4, 4]
// ---------------------------------------------------------------------------
__global__ __launch_bounds__(256)
void convt_k4s2(const float* __restrict__ in, const float* __restrict__ w,
                float* __restrict__ out, int B, int Cin, int Cout, int Hin, int Win) {
  const int Hout = Hin * 2, Wout = Win * 2;
  long i = (long)blockIdx.x * blockDim.x + threadIdx.x;
  long total = (long)B * Cout * Hout * Wout;
  if (i >= total) return;
  int ow = (int)(i % Wout); long t = i / Wout;
  int oh = (int)(t % Hout); t /= Hout;
  int co = (int)(t % Cout); int b = (int)(t / Cout);
  const size_t ipitch = (size_t)Hin * Win;       // per-ci input stride
  const size_t wpitch = (size_t)Cout * 16;       // per-ci weight stride
  float acc = 0.f;
#pragma unroll
  for (int th = 0; th < 2; ++th) {
    int ih = ((oh + 1) >> 1) - th;  int kh = ((oh + 1) & 1) + 2 * th;
    if ((unsigned)ih >= (unsigned)Hin) continue;
#pragma unroll
    for (int tw = 0; tw < 2; ++tw) {
      int iw = ((ow + 1) >> 1) - tw; int kw = ((ow + 1) & 1) + 2 * tw;
      if ((unsigned)iw >= (unsigned)Win) continue;
      const float* ip = in + ((size_t)b * Cin * Hin + ih) * Win + iw;
      const float* wp = w + (size_t)co * 16 + kh * 4 + kw;
      for (int ci = 0; ci < Cin; ++ci)
        acc = fmaf(ip[ci * ipitch], wp[ci * wpitch], acc);
    }
  }
  out[i] = acc;
}

// convT1 variant: z_q gathered on the fly from codebook[idx] (Cin=512,Cout=32,Hin=32)
__global__ __launch_bounds__(256)
void convt_vq(const float* __restrict__ cb, const int* __restrict__ idx,
              const float* __restrict__ w, float* __restrict__ out, int B) {
  long i = (long)blockIdx.x * blockDim.x + threadIdx.x;
  long total = (long)B << 17;                    // B*32*64*64
  if (i >= total) return;
  int ow = (int)(i & 63), oh = (int)((i >> 6) & 63);
  int co = (int)((i >> 12) & 31), b = (int)(i >> 17);
  float acc = 0.f;
#pragma unroll
  for (int th = 0; th < 2; ++th) {
    int ih = ((oh + 1) >> 1) - th;  int kh = ((oh + 1) & 1) + 2 * th;
    if ((unsigned)ih >= 32u) continue;
#pragma unroll
    for (int tw = 0; tw < 2; ++tw) {
      int iw = ((ow + 1) >> 1) - tw; int kw = ((ow + 1) & 1) + 2 * tw;
      if ((unsigned)iw >= 32u) continue;
      int code = idx[((size_t)b << 10) + (ih << 5) + iw];
      const float* cp = cb + (size_t)code * 512;         // z_q(b,:,ih,iw)
      const float* wp = w + (size_t)co * 16 + kh * 4 + kw;
      for (int ci = 0; ci < 512; ++ci)
        acc = fmaf(cp[ci], wp[(size_t)ci * 512], acc);
    }
  }
  out[i] = acc;
}

// ---------------------------------------------------------------------------
// Host launcher
// ---------------------------------------------------------------------------
extern "C" void kernel_launch(void* const* d_in, const int* in_sizes, int n_in,
                              void* d_out, int out_size, void* d_ws, size_t ws_size,
                              hipStream_t stream) {
  const float* x    = (const float*)d_in[0];
  const float* w1   = (const float*)d_in[1];
  const float* g1   = (const float*)d_in[3];
  const float* be1  = (const float*)d_in[4];
  const float* w2   = (const float*)d_in[5];
  const float* g2   = (const float*)d_in[7];
  const float* be2  = (const float*)d_in[8];
  const float* w3   = (const float*)d_in[9];
  const float* g3   = (const float*)d_in[11];
  const float* be3  = (const float*)d_in[12];
  const float* dw1  = (const float*)d_in[13];
  const float* dg1  = (const float*)d_in[15];
  const float* dbe1 = (const float*)d_in[16];
  const float* dw2  = (const float*)d_in[17];
  const float* dg2  = (const float*)d_in[19];
  const float* dbe2 = (const float*)d_in[20];
  const float* dw3  = (const float*)d_in[21];
  const float* dg3  = (const float*)d_in[23];
  const float* dbe3 = (const float*)d_in[24];
  const float* cbf  = (const float*)d_in[25];

  // Workspace layout (lifetime-based reuse), ~353 MiB total
  const size_t NEED = 370423808ull;
  if (ws_size < NEED) return;
  char* base = (char*)d_ws;
  float* h1 = (float*)(base + 0);                          // 64x16x128x128
  float* h2 = (float*)(base + 67108864ull);                // 64x32x64x64
  float* ze = (float*)(base + 100663296ull);               // 64x512x32x32 (NCHW)
  float* G  = (float*)(base + 234881024ull);               // 65536x512
  char*  sm = base + 369098752ull;
  unsigned short* w3bf = (unsigned short*)(sm);
  unsigned short* cbbf = (unsigned short*)(sm + 524288);
  float* cbn   = (float*)(sm + 1048576);
  float* meanb = (float*)(sm + 1050624);
  float* varb  = (float*)(sm + 1052672);
  int*   idxb  = (int*)  (sm + 1054720);
  float* part  = (float*)(sm + 1316864);
  unsigned short* A3  = (unsigned short*)h1;   // reuse: h1 dead after conv2
  unsigned short* zbf = (unsigned short*)h1;   // reuse: A3 dead after conv3 GEMM
  float* d1 = G;                               // reuse: G dead after argmin
  float* d2 = (float*)((char*)G + 33554432ull);

  float* out   = (float*)d_out;                // z_d [64,3,256,256]
  float* out_i = out + 12582912;               // idx as float [65536]
  float* out_l = out + 12648448;               // loss scalar

  // ---------------- encoder ----------------
  conv_k4s2<<<65536, 256, 0, stream>>>(x, w1, h1, 64, 3, 16, 256, 256);
  bn_stats<<<16, 256, 0, stream>>>(h1, meanb, varb, 16, 16384, 64);
  bn_act<<<65536, 256, 0, stream>>>(h1, g1, be1, meanb, varb, 16, 16384, 0);

  conv_k4s2<<<32768, 256, 0, stream>>>(h1, w2, h2, 64, 16, 32, 128, 128);
  bn_stats<<<32, 256, 0, stream>>>(h2, meanb, varb, 32, 4096, 64);
  bn_act<<<32768, 256, 0, stream>>>(h2, g2, be2, meanb, varb, 32, 4096, 0);

  // conv3 as bf16 WMMA GEMM (im2col 65536x512 x W3^T 512x512 -> z_e NCHW)
  im2col3<<<16384, 256, 0, stream>>>(h2, A3);
  cvt_bf16<<<128, 256, 0, stream>>>(w3, w3bf);             // 512*512 elements
  {
    dim3 grid(512 / 64, 65536 / 128);
    gemm_bf16_wmma<<<grid, 256, 0, stream>>>(A3, w3bf, ze, 65536, 512, 1);
  }
  bn_stats<<<512, 256, 0, stream>>>(ze, meanb, varb, 512, 1024, 64);
  bn_act<<<131072, 256, 0, stream>>>(ze, g3, be3, meanb, varb, 512, 1024, 0);

  // ---------------- VQ ----------------
  ze_to_flat_bf16<<<16384, 256, 0, stream>>>(ze, zbf);
  cvt_bf16<<<128, 256, 0, stream>>>(cbf, cbbf);
  cb_norm<<<2, 256, 0, stream>>>(cbf, cbn);
  {
    dim3 grid(512 / 64, 65536 / 128);
    gemm_bf16_wmma<<<grid, 256, 0, stream>>>(zbf, cbbf, G, 65536, 512, 0);
  }
  vq_argmin<<<8192, 256, 0, stream>>>(G, cbn, idxb, out_i);
  vq_loss<<<2048, 256, 0, stream>>>(ze, cbf, idxb, part);
  loss_final<<<1, 256, 0, stream>>>(part, out_l);

  // ---------------- decoder ----------------
  convt_vq<<<32768, 256, 0, stream>>>(cbf, idxb, dw1, d1, 64);
  bn_stats<<<32, 256, 0, stream>>>(d1, meanb, varb, 32, 4096, 64);
  bn_act<<<32768, 256, 0, stream>>>(d1, dg1, dbe1, meanb, varb, 32, 4096, 0);

  convt_k4s2<<<65536, 256, 0, stream>>>(d1, dw2, d2, 64, 32, 16, 64, 64);
  bn_stats<<<16, 256, 0, stream>>>(d2, meanb, varb, 16, 16384, 64);
  bn_act<<<65536, 256, 0, stream>>>(d2, dg2, dbe2, meanb, varb, 16, 16384, 0);

  convt_k4s2<<<49152, 256, 0, stream>>>(d2, dw3, out, 64, 16, 3, 128, 128);
  bn_stats<<<3, 256, 0, stream>>>(out, meanb, varb, 3, 65536, 64);
  bn_act<<<49152, 256, 0, stream>>>(out, dg3, dbe3, meanb, varb, 3, 65536, 1);
}
